// MultiTriplane_7447473292016
// MI455X (gfx1250) — matrix operand order repaired and checked by reading, not compiled, not run on hardware
//
#include <hip/hip_runtime.h>
#include <hip/hip_bf16.h>
#include <math.h>

// ---------------------------------------------------------------------------
// MI455X fused triplane-MLP occupancy loss.
// Precision: f16 WMMA (V_WMMA_F32_16X16X32_F16) with f32 accumulate.
// Weights pre-swizzled into B-fragment layout in d_ws; activations staged in
// LDS in A-fragment layout between layers.
// ---------------------------------------------------------------------------

typedef __attribute__((ext_vector_type(16))) _Float16 v16h;
typedef __attribute__((ext_vector_type(8)))  float    v8f;

#define HWSZ        (128 * 128)          // plane H*W
#define PLANE_ELEMS (32 * HWSZ)          // one plane [C=32,128,128]
#define FRAG_HALFS  512                  // one 16x32 (or 32x16) f16 fragment
// half-offsets inside the swizzled-weight region of d_ws:
#define OFF_BF 0                         // 4  frags (B_fourier 32x64)
#define OFF_W1 2048                      // 32 frags (W1 128x128)
#define OFF_W2 18432                     // 32 frags (W2 128x128)
#define OFF_W3 34816                     // 4  frags (W3 128x1, padded to 16 cols)
#define TOTAL_HALFS 36864

__device__ __forceinline__ v8f wmma16x16x32(v16h a, v16h b, v8f c) {
  return __builtin_amdgcn_wmma_f32_16x16x32_f16(
      /*neg_a=*/false, a, /*neg_b=*/false, b,
      /*c_mod=*/(short)0, c, /*reuse_a=*/false, /*reuse_b=*/false);
}

// ---------------------------------------------------------------------------
// Kernel 0: swizzle weights into per-lane B-fragment layout (f16) + zero accums
// B fragment (32x16 KxN), this file's k order (A matches): lane holds column n = lane&15;
// lanes 0-15 hold K=0..15, lanes 16-31 hold K=16..31, half index e == K%16.
// ---------------------------------------------------------------------------
__global__ __launch_bounds__(256) void swizzle_weights(
    const float* __restrict__ Bf, const float* __restrict__ W1,
    const float* __restrict__ W2, const float* __restrict__ W3,
    _Float16* __restrict__ hbase, float* __restrict__ accum) {
  int tid = blockIdx.x * 256 + threadIdx.x;
  if (tid < 2) accum[tid] = 0.0f;
  if (tid >= TOTAL_HALFS) return;
  float val;
  if (tid < OFF_W1) {                       // B_fourier [32,64]
    int frag = tid >> 9, rem = tid & 511;
    int l = rem >> 4, e = rem & 15;
    int K = e + ((l >> 4) << 4);
    int n = (frag << 4) | (l & 15);
    val = Bf[K * 64 + n];
  } else if (tid < OFF_W2) {                // W1 [128,128]
    int j = tid - OFF_W1;
    int frag = j >> 9, rem = j & 511;
    int kc = frag >> 3, nt = frag & 7;
    int l = rem >> 4, e = rem & 15;
    int K = kc * 32 + e + ((l >> 4) << 4);
    int n = nt * 16 + (l & 15);
    val = W1[K * 128 + n];
  } else if (tid < OFF_W3) {                // W2 [128,128]
    int j = tid - OFF_W2;
    int frag = j >> 9, rem = j & 511;
    int kc = frag >> 3, nt = frag & 7;
    int l = rem >> 4, e = rem & 15;
    int K = kc * 32 + e + ((l >> 4) << 4);
    int n = nt * 16 + (l & 15);
    val = W2[K * 128 + n];
  } else {                                  // W3 [128,1] padded to 16 columns
    int j = tid - OFF_W3;
    int kc = j >> 9, rem = j & 511;
    int l = rem >> 4, e = rem & 15;
    int K = kc * 32 + e + ((l >> 4) << 4);
    int n = l & 15;
    val = (n == 0) ? W3[K] : 0.0f;
  }
  hbase[tid] = (_Float16)val;
}

// ---------------------------------------------------------------------------
// Store one activation value h[row, K] into the per-wave LDS staging buffer,
// laid out as a 16x32 A fragment per K-chunk in this file's k order (B matches):
//   lane group = bit4 of K%32 ; half index e = K%32 & 15
// ---------------------------------------------------------------------------
__device__ __forceinline__ void store_h(_Float16* st, int row, int K, float v) {
  int kc = K >> 5, kl = K & 31;
  int l  = row + ((kl >> 4) << 4);
  int e  = kl & 15;
  st[kc * FRAG_HALFS + l * 16 + e] = (_Float16)v;
}

// ---------------------------------------------------------------------------
// Full pred_occ for a 16-point tile handled by one wave.
// pred[v] holds the logit for row (v + 8*(lane>>4)) on lanes with lane%16==0.
// ---------------------------------------------------------------------------
__device__ __forceinline__ void eval_points(
    const float* __restrict__ pl, const _Float16* __restrict__ hbase,
    const float* __restrict__ cptr, const float* __restrict__ nptr,
    int base, int lane, _Float16* st,
    const float* b1v, const float* b2v, float b3v, float pred[8]) {
  const int m = lane & 15, hi = lane >> 4;
  const int pt = base + m;
  float c0 = cptr[pt * 3 + 0], c1 = cptr[pt * 3 + 1], c2 = cptr[pt * 3 + 2];
  if (nptr) {
    c0 += nptr[pt * 3 + 0];
    c1 += nptr[pt * 3 + 1];
    c2 += nptr[pt * 3 + 2];
  }
  // plane coordinate pairs: (x,y) = (c0,c1), (c1,c2), (c0,c2)
  float ux[3] = {c0, c1, c0};
  float uy[3] = {c1, c2, c2};
  int   off00[3], off01[3], off10[3], off11[3];
  float w00[3], w01[3], w10[3], w11[3];
#pragma unroll
  for (int pi = 0; pi < 3; ++pi) {
    float fx = (ux[pi] + 1.0f) * 0.5f * 127.0f;
    float fy = (uy[pi] + 1.0f) * 0.5f * 127.0f;
    float fx0 = floorf(fx), fy0 = floorf(fy);
    float wx1 = fx - fx0, wy1 = fy - fy0;
    float wx0 = 1.0f - wx1, wy0 = 1.0f - wy1;
    int ix0 = (int)fx0, iy0 = (int)fy0;
    int ix1 = ix0 + 1, iy1 = iy0 + 1;
    float vx0 = ((unsigned)ix0 < 128u) ? 1.0f : 0.0f;
    float vx1 = ((unsigned)ix1 < 128u) ? 1.0f : 0.0f;
    float vy0 = ((unsigned)iy0 < 128u) ? 1.0f : 0.0f;
    float vy1 = ((unsigned)iy1 < 128u) ? 1.0f : 0.0f;
    int cx0 = min(max(ix0, 0), 127), cx1 = min(max(ix1, 0), 127);
    int cy0 = min(max(iy0, 0), 127), cy1 = min(max(iy1, 0), 127);
    w00[pi] = wy0 * wx0 * vy0 * vx0;
    w01[pi] = wy0 * wx1 * vy0 * vx1;
    w10[pi] = wy1 * wx0 * vy1 * vx0;
    w11[pi] = wy1 * wx1 * vy1 * vx1;
    off00[pi] = cy0 * 128 + cx0;
    off01[pi] = cy0 * 128 + cx1;
    off10[pi] = cy1 * 128 + cx0;
    off11[pi] = cy1 * 128 + cx1;
  }
  // Gather this lane's A-fragment slots directly (this file's contiguous k order, B matches):
  //   channel K = e + 16*hi
  v16h a;
#pragma unroll
  for (int e = 0; e < 16; ++e) {
    int K = e + 16 * hi;
    float f = 0.0f;
#pragma unroll
    for (int pi = 0; pi < 3; ++pi) {
      const float* P = pl + pi * PLANE_ELEMS + (K << 14);
      f += w00[pi] * P[off00[pi]] + w01[pi] * P[off01[pi]] +
           w10[pi] * P[off10[pi]] + w11[pi] * P[off11[pi]];
    }
    a[e] = (_Float16)(f * 6.283185307179586f);  // fold 2*pi into feats
  }
  // ---- Layer 0: proj = (2pi*feats) @ B_fourier  ([16,32]x[32,64]) ----
  v8f p4[4];
#pragma unroll
  for (int nt = 0; nt < 4; ++nt) {
    v16h b = *(const v16h*)(hbase + OFF_BF + nt * FRAG_HALFS + lane * 16);
    v8f z{};
    p4[nt] = wmma16x16x32(a, b, z);
  }
  // ---- stage h = [sin(proj), cos(proj)]  ->  A-fragment layout in LDS ----
#pragma unroll
  for (int nt = 0; nt < 4; ++nt)
#pragma unroll
    for (int v = 0; v < 8; ++v) {
      float p = p4[nt][v];
      int row = v + 8 * hi;
      int n = nt * 16 + m;
      store_h(st, row, n, __sinf(p));
      store_h(st, row, n + 64, __cosf(p));
    }
  asm volatile("s_wait_dscnt 0" ::: "memory");
  v16h A4[4];
#pragma unroll
  for (int kc = 0; kc < 4; ++kc)
    A4[kc] = *(const v16h*)(st + kc * FRAG_HALFS + lane * 16);
  // ---- Layer 1: h @ W1 ([16,128]x[128,128]) ----
  v8f acc[8];
#pragma unroll
  for (int nt = 0; nt < 8; ++nt) {
    v8f c{};
#pragma unroll
    for (int kc = 0; kc < 4; ++kc) {
      v16h b = *(const v16h*)(hbase + OFF_W1 + (kc * 8 + nt) * FRAG_HALFS +
                              lane * 16);
      c = wmma16x16x32(A4[kc], b, c);
    }
    acc[nt] = c;
  }
  asm volatile("s_wait_dscnt 0" ::: "memory");
#pragma unroll
  for (int nt = 0; nt < 8; ++nt)
#pragma unroll
    for (int v = 0; v < 8; ++v)
      store_h(st, v + 8 * hi, nt * 16 + m, fmaxf(acc[nt][v] + b1v[nt], 0.0f));
  asm volatile("s_wait_dscnt 0" ::: "memory");
#pragma unroll
  for (int kc = 0; kc < 4; ++kc)
    A4[kc] = *(const v16h*)(st + kc * FRAG_HALFS + lane * 16);
  // ---- Layer 2: @ W2 ----
#pragma unroll
  for (int nt = 0; nt < 8; ++nt) {
    v8f c{};
#pragma unroll
    for (int kc = 0; kc < 4; ++kc) {
      v16h b = *(const v16h*)(hbase + OFF_W2 + (kc * 8 + nt) * FRAG_HALFS +
                              lane * 16);
      c = wmma16x16x32(A4[kc], b, c);
    }
    acc[nt] = c;
  }
  asm volatile("s_wait_dscnt 0" ::: "memory");
#pragma unroll
  for (int nt = 0; nt < 8; ++nt)
#pragma unroll
    for (int v = 0; v < 8; ++v)
      store_h(st, v + 8 * hi, nt * 16 + m, fmaxf(acc[nt][v] + b2v[nt], 0.0f));
  asm volatile("s_wait_dscnt 0" ::: "memory");
#pragma unroll
  for (int kc = 0; kc < 4; ++kc)
    A4[kc] = *(const v16h*)(st + kc * FRAG_HALFS + lane * 16);
  // ---- Layer 3: @ W3 (padded to 16 cols; only column 0 is real) ----
  v8f c3{};
#pragma unroll
  for (int kc = 0; kc < 4; ++kc) {
    v16h b = *(const v16h*)(hbase + OFF_W3 + kc * FRAG_HALFS + lane * 16);
    c3 = wmma16x16x32(A4[kc], b, c3);
  }
#pragma unroll
  for (int v = 0; v < 8; ++v) pred[v] = c3[v] + b3v;
}

// ---------------------------------------------------------------------------
// Kernel 1: fused forward + loss partial reduction. One wave per 16-pt tile.
// ---------------------------------------------------------------------------
__global__ __launch_bounds__(256) void fused_forward(
    const int* __restrict__ obj_idx, const float* __restrict__ coords,
    const float* __restrict__ gt, const float* __restrict__ rc,
    const float* __restrict__ noise, const float* __restrict__ planes,
    const _Float16* __restrict__ hbase, const float* __restrict__ b1,
    const float* __restrict__ b2, const float* __restrict__ b3,
    float* __restrict__ accum, int ntiles) {
  __shared__ _Float16 stage_all[8 * 4 * FRAG_HALFS];  // 32 KB, 4 KB per wave
  const int tid = threadIdx.x;
  const int wid = tid >> 5;
  const int lane = tid & 31;
  const int tile = blockIdx.x * 8 + wid;
  if (tile >= ntiles) return;
  _Float16* st = &stage_all[wid * 4 * FRAG_HALFS];
  const int m = lane & 15, hi = lane >> 4;
  const int base = tile * 16;
  float b1v[8], b2v[8];
#pragma unroll
  for (int nt = 0; nt < 8; ++nt) {
    b1v[nt] = b1[nt * 16 + m];
    b2v[nt] = b2[nt * 16 + m];
  }
  const float b3v = b3[0];
  const float* pl = planes + (size_t)(3 * obj_idx[0]) * PLANE_ELEMS;

  float pred[8], predR[8];
  float bce = 0.0f, edr = 0.0f;
  const float g = (m == 0) ? 1.0f : 0.0f;  // only column n==0 holds logits

  // eval 0: coords -> BCE-with-logits partial (numerically stable form)
  eval_points(pl, hbase, coords, nullptr, base, lane, st, b1v, b2v, b3v, pred);
#pragma unroll
  for (int v = 0; v < 8; ++v) {
    float x = pred[v];
    float y = gt[base + v + 8 * hi];
    bce += g * (fmaxf(x, 0.0f) - x * y + log1pf(__expf(-fabsf(x))));
  }
  // eval 1 & 2: rand / rand+noise -> EDR partial
  eval_points(pl, hbase, rc, nullptr, base, lane, st, b1v, b2v, b3v, predR);
  eval_points(pl, hbase, rc, noise, base, lane, st, b1v, b2v, b3v, pred);
#pragma unroll
  for (int v = 0; v < 8; ++v) {
    float d = predR[v] - pred[v];
    edr += g * d * d;
  }
  // wave32 reduction
#pragma unroll
  for (int off = 16; off >= 1; off >>= 1) {
    bce += __shfl_xor(bce, off, 32);
    edr += __shfl_xor(edr, off, 32);
  }
  if (lane == 0) {
    atomicAdd(&accum[0], bce);
    atomicAdd(&accum[1], edr);
  }
}

// ---------------------------------------------------------------------------
// Kernel 2: finalize -> d_out = {total, naive, edr}
// ---------------------------------------------------------------------------
__global__ void finalize_loss(const float* __restrict__ accum,
                              float* __restrict__ out, float invN) {
  if (threadIdx.x == 0) {
    float naive = accum[0] * invN;
    float edr = accum[1] * invN * 0.3f;
    out[0] = naive + edr;
    out[1] = naive;
    out[2] = edr;
  }
}

extern "C" void kernel_launch(void* const* d_in, const int* in_sizes, int n_in,
                              void* d_out, int out_size, void* d_ws,
                              size_t ws_size, hipStream_t stream) {
  (void)n_in; (void)out_size; (void)ws_size;
  const int*   obj    = (const int*)d_in[0];
  const float* coords = (const float*)d_in[1];
  const float* gt     = (const float*)d_in[2];
  const float* rc     = (const float*)d_in[3];
  const float* noise  = (const float*)d_in[4];
  const float* planes = (const float*)d_in[5];
  const float* Bf     = (const float*)d_in[6];
  const float* W1     = (const float*)d_in[7];
  const float* b1     = (const float*)d_in[8];
  const float* W2     = (const float*)d_in[9];
  const float* b2     = (const float*)d_in[10];
  const float* W3     = (const float*)d_in[11];
  const float* b3     = (const float*)d_in[12];

  const int N = in_sizes[1] / 3;  // 524288
  float*     accum = (float*)d_ws;                      // [2] loss partials
  _Float16*  hbase = (_Float16*)((char*)d_ws + 256);    // swizzled f16 weights

  swizzle_weights<<<(TOTAL_HALFS + 255) / 256, 256, 0, stream>>>(
      Bf, W1, W2, W3, hbase, accum);

  const int ntiles = N / 16;
  const int blocks = (ntiles + 7) / 8;  // 8 waves (tiles) per 256-thread block
  fused_forward<<<blocks, 256, 0, stream>>>(obj, coords, gt, rc, noise, planes,
                                            hbase, b1, b2, b3, accum, ntiles);

  finalize_loss<<<1, 32, 0, stream>>>(accum, (float*)d_out, 1.0f / (float)N);
}
